// DeformableConvBlock_47485158425306
// MI455X (gfx1250) — matrix operand order repaired and checked by reading, not compiled, not run on hardware
//
#include <hip/hip_runtime.h>
#include <hip/hip_bf16.h>
#include <stdint.h>

typedef __attribute__((ext_vector_type(16))) __bf16 v16bf;
typedef __attribute__((ext_vector_type(8)))  float   v8f;

#define C_IN   64
#define C_OUT  64
#define K2     9
#define HH     128
#define WW     128
#define BB     4
#define NPIX   (BB*HH*WW)        // 65536
#define KDIM   (C_IN*K2)         // 576
#define LDSROW 584               // ushorts per LDS A-row: 576 + 8 pad (1168 B, 16B aligned)
#define EPSBN  1e-5f

static __device__ __forceinline__ unsigned short f2bf(float f) {
  union { float f; uint32_t u; } v; v.f = f;
  uint32_t u = v.u;
  uint32_t r = (u + 0x7FFFu + ((u >> 16) & 1u)) >> 16;   // RNE fp32->bf16
  return (unsigned short)r;
}

// ---- Stage 1: dcn weights fp32 -> bf16, permuted to ck = k*64 + c ----------
__global__ void k_convert_w(const float* __restrict__ w,
                            unsigned short* __restrict__ wb, int n) {
  int i = blockIdx.x * 256 + threadIdx.x;
  if (i >= n) return;
  int o = i / KDIM;
  int r = i % KDIM;           // r = k*64 + c
  int k = r >> 6;
  int c = r & 63;
  wb[i] = f2bf(w[o * KDIM + c * K2 + k]);
}

// ---- Stage 2: offset conv + bilinear sampling -> im2col bf16 ---------------
// Sb layout: [n][ck], n = b*16384 + y*128 + x, ck = k*64 + c  (row = 1152 B)
__global__ void k_offset_sample(const float* __restrict__ X,
                                const float* __restrict__ OW,   // (18,64,3,3)
                                const float* __restrict__ OB,   // (18,)
                                unsigned short* __restrict__ Sb) {
  int n  = blockIdx.x * 256 + threadIdx.x;
  int b  = n >> 14;
  int hw = n & 16383;
  int y  = hw >> 7;
  int x  = hw & 127;

  float off[18];
#pragma unroll
  for (int j = 0; j < 18; ++j) off[j] = OB[j];

  const float* xb0 = X + (size_t)b * (C_IN * HH * WW);

  // direct 3x3 conv, zero padding (weights via uniform/scalar loads)
  for (int ki = 0; ki < 3; ++ki) {
    int yy = y + ki - 1;
    if (yy < 0 || yy >= HH) continue;
    for (int kj = 0; kj < 3; ++kj) {
      int xx = x + kj - 1;
      if (xx < 0 || xx >= WW) continue;
      const float* xp = xb0 + yy * WW + xx;
      const float* wp = OW + ki * 3 + kj;
      for (int c = 0; c < C_IN; ++c) {
        float xv = xp[c * HH * WW];
#pragma unroll
        for (int j = 0; j < 18; ++j)
          off[j] = fmaf(xv, wp[(j * C_IN + c) * K2], off[j]);
      }
    }
  }

  unsigned short* srow = Sb + (size_t)n * KDIM;
#pragma unroll
  for (int k = 0; k < 9; ++k) {
    int ki = k / 3, kj = k % 3;
    float py = off[2*k]     + (float)(y + ki - 1);
    float px = off[2*k + 1] + (float)(x + kj - 1);
    float y0f = floorf(py), x0f = floorf(px);
    float wy = py - y0f, wx = px - x0f;
    int iy0 = (int)y0f, ix0 = (int)x0f;
    int iy1 = iy0 + 1,  ix1 = ix0 + 1;
    float vy0 = (iy0 >= 0 && iy0 < HH) ? 1.f : 0.f;
    float vy1 = (iy1 >= 0 && iy1 < HH) ? 1.f : 0.f;
    float vx0 = (ix0 >= 0 && ix0 < WW) ? 1.f : 0.f;
    float vx1 = (ix1 >= 0 && ix1 < WW) ? 1.f : 0.f;
    int cy0 = min(max(iy0, 0), HH - 1), cy1 = min(max(iy1, 0), HH - 1);
    int cx0 = min(max(ix0, 0), WW - 1), cx1 = min(max(ix1, 0), WW - 1);
    int o00 = cy0 * WW + cx0, o01 = cy0 * WW + cx1;
    int o10 = cy1 * WW + cx0, o11 = cy1 * WW + cx1;
    float w00 = (1.f - wy) * (1.f - wx) * vy0 * vx0;
    float w01 = (1.f - wy) * wx         * vy0 * vx1;
    float w10 = wy         * (1.f - wx) * vy1 * vx0;
    float w11 = wy         * wx         * vy1 * vx1;

    // 64 channels for this k, packed 8 bf16 per b128 store (128B aligned base)
    uint32_t* dst = (uint32_t*)(srow + k * 64);
    for (int c8 = 0; c8 < 64; c8 += 8) {
      uint4 pk;
      uint32_t tmp[4];
#pragma unroll
      for (int j = 0; j < 4; ++j) {
        const float* xa = xb0 + (c8 + 2*j)     * (HH * WW);
        const float* xc = xb0 + (c8 + 2*j + 1) * (HH * WW);
        float s0 = w00*xa[o00] + w01*xa[o01] + w10*xa[o10] + w11*xa[o11];
        float s1 = w00*xc[o00] + w01*xc[o01] + w10*xc[o10] + w11*xc[o11];
        tmp[j] = (uint32_t)f2bf(s0) | ((uint32_t)f2bf(s1) << 16);
      }
      pk.x = tmp[0]; pk.y = tmp[1]; pk.z = tmp[2]; pk.w = tmp[3];
      *(uint4*)(dst + c8 / 2) = pk;
    }
  }
}

// ---- Stage 3: WMMA GEMM  out[o][n] = Wb[o][ck] * Sb[n][ck] -----------------
// Block = 8 waves. A (64x576 bf16) staged in LDS once; per wave 64(M)x32(N).
__global__ void __launch_bounds__(256) k_gemm(const unsigned short* __restrict__ Wb,
                                              const unsigned short* __restrict__ Sb,
                                              float* __restrict__ Of) {
  extern __shared__ unsigned short lA[];   // 64 * LDSROW ushorts = 74,752 B

  // cooperative stage: 64 rows * 72 uint4
  for (int i = threadIdx.x; i < C_OUT * (KDIM / 8); i += 256) {
    int row = i / (KDIM / 8);
    int col = i - row * (KDIM / 8);
    *(uint4*)&lA[row * LDSROW + col * 8] = *(const uint4*)&Wb[row * KDIM + col * 8];
  }
  __syncthreads();

  int lane  = threadIdx.x & 31;
  int wave  = threadIdx.x >> 5;
  int n0    = (blockIdx.x * 8 + wave) * 32;
  int lrow  = lane & 15;
  int lhalf = lane >> 4;

  v8f acc[8];
#pragma unroll
  for (int t = 0; t < 8; ++t)
#pragma unroll
    for (int i = 0; i < 8; ++i) acc[t][i] = 0.f;

  // B fragments: lane holds K = lhalf*16 + 0..15 of column n (contig 32B)
  const unsigned short* bbase0 = Sb + (size_t)(n0 + lrow) * KDIM + lhalf * 16;
  const unsigned short* bbase1 = bbase0 + (size_t)16 * KDIM;
  // A fragments from LDS: lane holds M = lrow, K = lhalf*16 + 0..15
  const unsigned short* labase = lA + lrow * LDSROW + lhalf * 16;

  union Frag { v16bf v; uint4 q[2]; };

  Frag b0c, b1c;
  b0c.q[0] = *(const uint4*)(bbase0);     b0c.q[1] = *(const uint4*)(bbase0 + 8);
  b1c.q[0] = *(const uint4*)(bbase1);     b1c.q[1] = *(const uint4*)(bbase1 + 8);

  for (int k = 0; k < KDIM; k += 32) {
    int kn = k + 32; if (kn >= KDIM) kn = 0;   // harmless wrap prefetch
    Frag b0n, b1n;
    b0n.q[0] = *(const uint4*)(bbase0 + kn);  b0n.q[1] = *(const uint4*)(bbase0 + kn + 8);
    b1n.q[0] = *(const uint4*)(bbase1 + kn);  b1n.q[1] = *(const uint4*)(bbase1 + kn + 8);
    __builtin_prefetch(bbase0 + k + 128, 0, 1);   // global_prefetch_b8 on B stream

    Frag a[4];
#pragma unroll
    for (int mt = 0; mt < 4; ++mt) {
      const unsigned short* ap = labase + mt * 16 * LDSROW + k;
      a[mt].q[0] = *(const uint4*)(ap);
      a[mt].q[1] = *(const uint4*)(ap + 8);
    }
#pragma unroll
    for (int mt = 0; mt < 4; ++mt) {
      acc[mt]     = __builtin_amdgcn_wmma_f32_16x16x32_bf16(false, a[mt].v, false, b0c.v, (short)0, acc[mt],     false, false);
      acc[mt + 4] = __builtin_amdgcn_wmma_f32_16x16x32_bf16(false, a[mt].v, false, b1c.v, (short)0, acc[mt + 4], false, false);
    }
    b0c = b0n; b1c = b1n;
  }

  // C/D layout: VGPR r -> M = lhalf*8 + r, lane&15 -> N
  float* orow0 = Of + n0 + lrow;
  float* orow1 = orow0 + 16;
#pragma unroll
  for (int r = 0; r < 8; ++r) {
    int m0 = lhalf * 8 + r;
#pragma unroll
    for (int mt = 0; mt < 4; ++mt) {
      orow0[(size_t)(m0 + 16 * mt) * NPIX] = acc[mt][r];
      orow1[(size_t)(m0 + 16 * mt) * NPIX] = acc[mt + 4][r];
    }
  }
}

// ---- Stage 4: per-channel mean / rstd --------------------------------------
__global__ void k_bn_stats(const float* __restrict__ Of, float* __restrict__ stats) {
  __shared__ float ssum[256], ssq[256];
  int o = blockIdx.x;
  const float* p = Of + (size_t)o * NPIX;
  float s = 0.f, q = 0.f;
  for (int i = threadIdx.x; i < NPIX; i += 256) { float v = p[i]; s += v; q += v * v; }
  ssum[threadIdx.x] = s; ssq[threadIdx.x] = q;
  __syncthreads();
  for (int st = 128; st > 0; st >>= 1) {
    if (threadIdx.x < st) {
      ssum[threadIdx.x] += ssum[threadIdx.x + st];
      ssq[threadIdx.x]  += ssq[threadIdx.x + st];
    }
    __syncthreads();
  }
  if (threadIdx.x == 0) {
    float mean = ssum[0] / (float)NPIX;
    float var  = ssq[0] / (float)NPIX - mean * mean;
    stats[o * 2]     = mean;
    stats[o * 2 + 1] = rsqrtf(var + EPSBN);
  }
}

// ---- Stage 5: BN + gamma/beta + ReLU, scatter to NCHW ----------------------
__global__ void k_bn_apply(const float* __restrict__ Of, const float* __restrict__ stats,
                           const float* __restrict__ gamma, const float* __restrict__ beta,
                           float* __restrict__ out) {
  int idx = blockIdx.x * 256 + threadIdx.x;   // o*NPIX + n
  int o = idx >> 16;
  int n = idx & (NPIX - 1);
  float mean = stats[o * 2], rstd = stats[o * 2 + 1];
  float v = (Of[idx] - mean) * rstd * gamma[o] + beta[o];
  v = fmaxf(v, 0.f);
  int b = n >> 14, hw = n & 16383;
  out[((size_t)b * C_OUT + o) * (HH * WW) + hw] = v;
}

extern "C" void kernel_launch(void* const* d_in, const int* in_sizes, int n_in,
                              void* d_out, int out_size, void* d_ws, size_t ws_size,
                              hipStream_t stream) {
  const float* x        = (const float*)d_in[0];
  const float* offset_w = (const float*)d_in[1];
  const float* offset_b = (const float*)d_in[2];
  const float* dcn_w    = (const float*)d_in[3];
  const float* gamma    = (const float*)d_in[4];
  const float* beta     = (const float*)d_in[5];
  float* out = (float*)d_out;

  char* ws = (char*)d_ws;
  const size_t sb_bytes = (size_t)NPIX * KDIM * sizeof(unsigned short);   // 75.5 MB
  const size_t wb_bytes = (size_t)C_OUT * KDIM * sizeof(unsigned short);  // 73.7 KB
  unsigned short* Sb = (unsigned short*)(ws);
  unsigned short* Wb = (unsigned short*)(ws + sb_bytes);
  float* Of          = (float*)(ws + sb_bytes + wb_bytes);                // 16.8 MB
  float* stats       = (float*)(ws + sb_bytes + wb_bytes + (size_t)C_OUT * NPIX * sizeof(float));

  const int lds_bytes = C_OUT * LDSROW * sizeof(unsigned short);          // 74,752 B

  k_convert_w<<<(C_OUT * KDIM + 255) / 256, 256, 0, stream>>>(dcn_w, Wb, C_OUT * KDIM);
  k_offset_sample<<<NPIX / 256, 256, 0, stream>>>(x, offset_w, offset_b, Sb);
  k_gemm<<<(NPIX / 32) / 8, 256, lds_bytes, stream>>>(Wb, Sb, Of);
  k_bn_stats<<<C_OUT, 256, 0, stream>>>(Of, stats);
  k_bn_apply<<<(C_OUT * NPIX) / 256, 256, 0, stream>>>(Of, stats, gamma, beta, out);
}
